// SwinTransformerBlock_59751585022210
// MI455X (gfx1250) — matrix operand order repaired and checked by reading, not compiled, not run on hardware
//
#include <hip/hip_runtime.h>
#include <hip/hip_bf16.h>
#include <math.h>

// Swin block for MI455X (gfx1250, wave32, WMMA 16x16x32 f16).
// Workspace requirement: ~521 MB (weights f16 + shared act region + qkv/mlp-hidden region).

#define Bn   32
#define Hh   56
#define Ww   56
#define Cc   512
#define NHh  16
#define HD   32
#define WSz  7
#define SSz  3
#define Nt   49          // tokens per window
#define NWw  64          // windows per image
#define TOK  (Bn * Hh * Ww)   // 100352

static constexpr float kScale = 0.1767766952966369f;  // 32^-0.5

typedef _Float16 v16h __attribute__((ext_vector_type(16)));
typedef _Float16 v8h  __attribute__((ext_vector_type(8)));
typedef float    v8f  __attribute__((ext_vector_type(8)));

// ---------------------------------------------------------------------------
// WMMA helpers (gfx1250 layouts)
// ---------------------------------------------------------------------------
__device__ inline v8f wmma_f32_f16(v16h a, v16h b, v8f c) {
  return __builtin_amdgcn_wmma_f32_16x16x32_f16(
      /*neg_a=*/false, a, /*neg_b=*/false, b,
      /*c_mod=*/(short)0, c, /*reuse_a=*/false, /*reuse_b=*/false);
}

// A fragment (16x32 f16): lane = M row (lane&15); K set depends on lane half.
// i<8  -> K = 8*half + i ; i>=8 -> K = 16 + 8*half + (i-8)   (two b128 LDS loads)
__device__ inline v16h load_a_frag(const _Float16* base, int stride, int row0,
                                   int k0, int lane) {
  int r = row0 + (lane & 15);
  int half = lane >> 4;
  const _Float16* p = base + (size_t)r * stride + k0;
  v8h lo = *(const v8h*)(p + 8 * half);
  v8h hi = *(const v8h*)(p + 16 + 8 * half);
  return __builtin_shufflevector(lo, hi, 0, 1, 2, 3, 4, 5, 6, 7,
                                 8, 9, 10, 11, 12, 13, 14, 15);
}

// B fragment (32x16 f16) from a [n][k] row-major LDS tile:
// lane = N col (lane&15); i<8 -> K = 8*half + i ; i>=8 -> K = 16 + 8*half + (i-8)   (two b128 LDS loads)
__device__ inline v16h load_b_frag(const _Float16* base, int stride, int col0,
                                   int k0, int lane) {
  int c = col0 + (lane & 15);
  int kh = lane >> 4;
  const _Float16* p = base + (size_t)c * stride + k0;
  v8h lo = *(const v8h*)(p + 8 * kh);
  v8h hi = *(const v8h*)(p + 16 + 8 * kh);
  return __builtin_shufflevector(lo, hi, 0, 1, 2, 3, 4, 5, 6, 7,
                                 8, 9, 10, 11, 12, 13, 14, 15);
}

// ---------------------------------------------------------------------------
// fp32 -> fp16 weight conversion
// ---------------------------------------------------------------------------
__global__ __launch_bounds__(256) void cvt_kernel(const float* __restrict__ s,
                                                  _Float16* __restrict__ d,
                                                  int n) {
  int i = blockIdx.x * 256 + threadIdx.x;
  if (i < n) d[i] = (_Float16)s[i];
}

// ---------------------------------------------------------------------------
// LayerNorm (one block per token, 256 threads, C=512).
// remap=1: output row m is a windowed token (LN1 + cyclic shift + partition).
// remap=0: plain LN over token m (LN2).
// ---------------------------------------------------------------------------
__global__ __launch_bounds__(256) void ln_kernel(const float* __restrict__ x,
                                                 const float* __restrict__ g,
                                                 const float* __restrict__ bb,
                                                 _Float16* __restrict__ out,
                                                 int remap) {
  __shared__ float red[8];
  __shared__ float bc[2];
  int m = blockIdx.x;
  int tid = threadIdx.x;

  size_t srcRow;
  if (remap) {
    int bI  = m / 3136;            // 64 windows * 49 tokens = 3136 = H*W
    int r2  = m - bI * 3136;
    int wsp = r2 / Nt;
    int t   = r2 - wsp * Nt;
    int hp  = (wsp >> 3) * WSz + t / WSz;   // shifted-frame coords
    int wp  = (wsp & 7) * WSz + t % WSz;
    int hh  = hp + SSz; if (hh >= Hh) hh -= Hh;   // shifted[h'] = x[(h'+3)%56]
    int ww2 = wp + SSz; if (ww2 >= Ww) ww2 -= Ww;
    srcRow = (size_t)bI * 3136 + hh * Ww + ww2;
  } else {
    srcRow = (size_t)m;
  }

  const float* xr = x + srcRow * Cc;
  float a = xr[tid], b2 = xr[tid + 256];

  float s = a + b2;
  #pragma unroll
  for (int off = 16; off; off >>= 1) s += __shfl_xor(s, off);
  if ((tid & 31) == 0) red[tid >> 5] = s;
  __syncthreads();
  if (tid == 0) {
    float t2 = 0.f;
    #pragma unroll
    for (int i = 0; i < 8; ++i) t2 += red[i];
    bc[0] = t2 * (1.0f / Cc);
  }
  __syncthreads();
  float mu = bc[0];

  float da = a - mu, db = b2 - mu;
  float vsum = da * da + db * db;
  #pragma unroll
  for (int off = 16; off; off >>= 1) vsum += __shfl_xor(vsum, off);
  if ((tid & 31) == 0) red[tid >> 5] = vsum;
  __syncthreads();
  if (tid == 0) {
    float t2 = 0.f;
    #pragma unroll
    for (int i = 0; i < 8; ++i) t2 += red[i];
    bc[1] = t2 * (1.0f / Cc);
  }
  __syncthreads();
  float inv = rsqrtf(bc[1] + 1e-5f);

  out[(size_t)m * Cc + tid]       = (_Float16)(da * inv * g[tid] + bb[tid]);
  out[(size_t)m * Cc + tid + 256] = (_Float16)(db * inv * g[tid + 256] + bb[tid + 256]);
}

// ---------------------------------------------------------------------------
// Tiled WMMA GEMM: out[M x N] = A[M x K] (f16) @ Wt[N x K]^T (f16), f32 accum.
// Block tile 128x128, K-steps of 32, double-buffered LDS, 256 threads = 8
// waves, each wave owns 2 M-tiles x 4 N-tiles (8 WMMAs per K-step, 1 barrier).
// ---------------------------------------------------------------------------
#define EPI_QKV     0   // +bias; scale cols [0,512) by kScale; f16 store
#define EPI_OUTPROJ 2   // +bias; window->token scatter; += shortcut; f32 store
#define EPI_GELU    3   // +bias; exact GELU; f16 store
#define EPI_FC2     4   // +bias; f32 accumulate into outF (residual)

__global__ __launch_bounds__(256) void gemm_f16(
    const _Float16* __restrict__ A, const _Float16* __restrict__ Wt,
    const float* __restrict__ bias, int M, int Nn, int K, int epi,
    _Float16* __restrict__ outH, float* __restrict__ outF,
    const float* __restrict__ resid) {
  __shared__ _Float16 As[2][128][40];
  __shared__ _Float16 Bs[2][128][40];

  int tid = threadIdx.x, lane = tid & 31, wave = tid >> 5;
  int bm = blockIdx.x * 128, bn = blockIdx.y * 128;
  int wm = wave & 3;        // 4 M-groups of 2 tiles
  int wn = wave >> 2;       // 2 N-groups of 4 tiles

  v8f acc[2][4] = {};

  // Global staging: each thread owns 16 contiguous f16 of one row per matrix.
  int grow = tid >> 1;
  int gk = (tid & 1) * 16;
  const _Float16* Ag = &A[(size_t)(bm + grow) * K + gk];
  const _Float16* Bg = &Wt[(size_t)(bn + grow) * K + gk];

  // Prologue: stage K-step 0.
  v8h a0 = *(const v8h*)Ag;
  v8h a1 = *(const v8h*)(Ag + 8);
  v8h b0 = *(const v8h*)Bg;
  v8h b1 = *(const v8h*)(Bg + 8);
  *(v8h*)&As[0][grow][gk]     = a0;
  *(v8h*)&As[0][grow][gk + 8] = a1;
  *(v8h*)&Bs[0][grow][gk]     = b0;
  *(v8h*)&Bs[0][grow][gk + 8] = b1;
  __syncthreads();

  int steps = K >> 5;
  int cur = 0;
  for (int s = 0; s < steps; ++s) {
    // Issue next tile's global loads first; s_wait_loadcnt lands after WMMAs.
    if (s + 1 < steps) {
      const _Float16* An = Ag + (size_t)(s + 1) * 32;
      const _Float16* Bnxt = Bg + (size_t)(s + 1) * 32;
      a0 = *(const v8h*)An;
      a1 = *(const v8h*)(An + 8);
      b0 = *(const v8h*)Bnxt;
      b1 = *(const v8h*)(Bnxt + 8);
      if (s + 2 < steps) {             // deep prefetch -> global_prefetch_b8
        __builtin_prefetch(An + 32, 0, 0);
        __builtin_prefetch(Bnxt + 32, 0, 0);
      }
    }

    // Fragments + 8 WMMAs from the current LDS buffer.
    v16h af0 = load_a_frag(&As[cur][0][0], 40, (2 * wm) * 16, 0, lane);
    v16h af1 = load_a_frag(&As[cur][0][0], 40, (2 * wm + 1) * 16, 0, lane);
    v16h bf[4];
    #pragma unroll
    for (int j = 0; j < 4; ++j)
      bf[j] = load_b_frag(&Bs[cur][0][0], 40, (4 * wn + j) * 16, 0, lane);
    #pragma unroll
    for (int j = 0; j < 4; ++j) {
      acc[0][j] = wmma_f32_f16(af0, bf[j], acc[0][j]);
      acc[1][j] = wmma_f32_f16(af1, bf[j], acc[1][j]);
    }

    // Store next tile into the other buffer; one barrier per K-step.
    if (s + 1 < steps) {
      int nxt = cur ^ 1;
      *(v8h*)&As[nxt][grow][gk]     = a0;
      *(v8h*)&As[nxt][grow][gk + 8] = a1;
      *(v8h*)&Bs[nxt][grow][gk]     = b0;
      *(v8h*)&Bs[nxt][grow][gk + 8] = b1;
    }
    __syncthreads();
    cur ^= 1;
  }

  // Epilogue: C/D layout M = r + 8*(lane>>4), N = lane&15.
  int nl = lane & 15;
  int mh = (lane >> 4) * 8;
  int ncol[4];
  float bv[4];
  #pragma unroll
  for (int j = 0; j < 4; ++j) {
    ncol[j] = bn + (4 * wn + j) * 16 + nl;
    bv[j] = bias ? bias[ncol[j]] : 0.f;
  }
  #pragma unroll
  for (int i = 0; i < 2; ++i) {
    #pragma unroll
    for (int r = 0; r < 8; ++r) {
      int m = bm + (2 * wm + i) * 16 + mh + r;
      if (epi == EPI_OUTPROJ) {
        // windowed row -> original token (reverse shift), once per row
        int bI  = m / 3136;
        int r2  = m - bI * 3136;
        int wsp = r2 / Nt;
        int tk  = r2 - wsp * Nt;
        int hp  = (wsp >> 3) * WSz + tk / WSz;
        int wp  = (wsp & 7) * WSz + tk % WSz;
        int hh  = hp + SSz; if (hh >= Hh) hh -= Hh;
        int ww2 = wp + SSz; if (ww2 >= Ww) ww2 -= Ww;
        size_t tok = (size_t)bI * 3136 + hh * Ww + ww2;
        #pragma unroll
        for (int j = 0; j < 4; ++j) {
          float v = acc[i][j][r] + bv[j];
          outF[tok * Cc + ncol[j]] = v + resid[tok * Cc + ncol[j]];
        }
      } else if (epi == EPI_QKV) {
        #pragma unroll
        for (int j = 0; j < 4; ++j) {
          float v = acc[i][j][r] + bv[j];
          if (ncol[j] < Cc) v *= kScale;      // fold softmax scale into Q
          outH[(size_t)m * Nn + ncol[j]] = (_Float16)v;
        }
      } else if (epi == EPI_GELU) {
        #pragma unroll
        for (int j = 0; j < 4; ++j) {
          float v = acc[i][j][r] + bv[j];
          v = 0.5f * v * (1.0f + erff(v * 0.7071067811865475f));
          outH[(size_t)m * Nn + ncol[j]] = (_Float16)v;
        }
      } else {  // EPI_FC2: accumulate into residual held in outF
        #pragma unroll
        for (int j = 0; j < 4; ++j) {
          float v = acc[i][j][r] + bv[j];
          outF[(size_t)m * Cc + ncol[j]] += v;
        }
      }
    }
  }
}

// ---------------------------------------------------------------------------
// Windowed attention: one block (128 threads = 4 waves) per (window, head).
// S = (Q*scale) K^T + rp_bias + shift_mask ; softmax ; O = P V.
// N=49 padded to 64; HD=32 (one WMMA K-step for S, two for P@V).
// ---------------------------------------------------------------------------
__global__ __launch_bounds__(128) void attn_kernel(
    const _Float16* __restrict__ qkv,    // [100352][1536]  (q | k | v)
    const float* __restrict__ rpb,       // [169][16]
    _Float16* __restrict__ o) {          // [100352][512]
  __shared__ _Float16 Qs[64][40];
  __shared__ _Float16 Ks[64][40];
  __shared__ _Float16 Vt[32][72];        // V transposed: [d][token]
  __shared__ float    Sf[64][65];
  __shared__ _Float16 Ph[64][72];

  int wID  = blockIdx.x;       // 0..2047  (b*64 + spatial window)
  int head = blockIdx.y;       // 0..15
  int tid = threadIdx.x, lane = tid & 31, wave = tid >> 5;
  int wsp = wID & 63;
  int wh = wsp >> 3, wwc = wsp & 7;

  // Stage Q, K, V tiles (zero-pad tokens 49..63).
  for (int idx = tid; idx < 64 * 32; idx += 128) {
    int t = idx >> 5, d = idx & 31;
    _Float16 q = (_Float16)0, k = (_Float16)0, vv = (_Float16)0;
    if (t < Nt) {
      size_t row = (size_t)(wID * Nt + t) * 1536 + head * HD + d;
      q  = qkv[row];
      k  = qkv[row + 512];
      vv = qkv[row + 1024];
    }
    Qs[t][d] = q;
    Ks[t][d] = k;
    Vt[d][t] = vv;
  }
  __syncthreads();

  // S = Q K^T  (wave owns one 16-row tile, all 4 col tiles)
  int nl = lane & 15, half = lane >> 4;
  v16h aq = load_a_frag(&Qs[0][0], 40, wave * 16, 0, lane);
  #pragma unroll
  for (int ni = 0; ni < 4; ++ni) {
    v8f s = {};
    v16h bk = load_b_frag(&Ks[0][0], 40, ni * 16, 0, lane);
    s = wmma_f32_f16(aq, bk, s);
    int n = ni * 16 + nl;
    #pragma unroll
    for (int r = 0; r < 8; ++r) {
      int m = wave * 16 + 8 * half + r;
      float v = s[r];
      if (m < Nt && n < Nt) {
        int qh = m / WSz, qw = m % WSz;
        int kh = n / WSz, kw = n % WSz;
        v += rpb[((qh - kh + 6) * 13 + (qw - kw + 6)) * NHh + head];
        // shifted-window mask: region labels on the shifted frame
        int ghq = wh * WSz + qh, gwq = wwc * WSz + qw;
        int ghk = wh * WSz + kh, gwk = wwc * WSz + kw;
        int rq = ((ghq < 49) ? 0 : (ghq < 53 ? 1 : 2)) * 3 +
                 ((gwq < 49) ? 0 : (gwq < 53 ? 1 : 2));
        int rk = ((ghk < 49) ? 0 : (ghk < 53 ? 1 : 2)) * 3 +
                 ((gwk < 49) ? 0 : (gwk < 53 ? 1 : 2));
        if (rq != rk) v -= 100.0f;
      } else {
        v = -1e30f;   // pad columns never win softmax
      }
      Sf[m][n] = v;
    }
  }
  __syncthreads();

  // Row softmax; write P (f16, zero-padded to K=64).
  if (tid < 64) {
    int r = tid;
    float mx = -1e30f;
    for (int c2 = 0; c2 < Nt; ++c2) mx = fmaxf(mx, Sf[r][c2]);
    float sum = 0.f;
    for (int c2 = 0; c2 < Nt; ++c2) {
      float e = __expf(Sf[r][c2] - mx);
      Sf[r][c2] = e;
      sum += e;
    }
    float invs = 1.0f / sum;
    for (int c2 = 0; c2 < 64; ++c2)
      Ph[r][c2] = (c2 < Nt) ? (_Float16)(Sf[r][c2] * invs) : (_Float16)0;
  }
  __syncthreads();

  // O = P @ V  (wave owns one 16-row tile x 2 col tiles, K = 64 in 2 steps)
  v8f oacc0 = {}; v8f oacc1 = {};
  #pragma unroll
  for (int ks = 0; ks < 2; ++ks) {
    v16h pa = load_a_frag(&Ph[0][0], 72, wave * 16, ks * 32, lane);
    v16h v0 = load_b_frag(&Vt[0][0], 72, 0,  ks * 32, lane);
    v16h v1 = load_b_frag(&Vt[0][0], 72, 16, ks * 32, lane);
    oacc0 = wmma_f32_f16(pa, v0, oacc0);
    oacc1 = wmma_f32_f16(pa, v1, oacc1);
  }
  #pragma unroll
  for (int r = 0; r < 8; ++r) {
    int m = wave * 16 + 8 * half + r;
    if (m < Nt) {
      size_t base = (size_t)(wID * Nt + m) * Cc + head * HD;
      o[base + nl]      = (_Float16)oacc0[r];
      o[base + 16 + nl] = (_Float16)oacc1[r];
    }
  }
}

// ---------------------------------------------------------------------------
// Host-side launch
// ---------------------------------------------------------------------------
extern "C" void kernel_launch(void* const* d_in, const int* in_sizes, int n_in,
                              void* d_out, int out_size, void* d_ws,
                              size_t ws_size, hipStream_t stream) {
  const float* x      = (const float*)d_in[0];
  const float* n1w    = (const float*)d_in[1];
  const float* n1b    = (const float*)d_in[2];
  const float* qkv_w  = (const float*)d_in[3];
  const float* qkv_b  = (const float*)d_in[4];
  const float* rpb    = (const float*)d_in[5];
  const float* out_w  = (const float*)d_in[6];
  const float* out_b  = (const float*)d_in[7];
  const float* n2w    = (const float*)d_in[8];
  const float* n2b    = (const float*)d_in[9];
  const float* fc1_w  = (const float*)d_in[10];
  const float* fc1_b  = (const float*)d_in[11];
  const float* fc2_w  = (const float*)d_in[12];
  const float* fc2_b  = (const float*)d_in[13];
  float* out = (float*)d_out;

  // Workspace carve-up (aligned to 256B).
  char* w = (char*)d_ws;
  size_t off = 0;
  auto carve = [&](size_t bytes) {
    void* p = w + off;
    off = (off + bytes + 255) & ~(size_t)255;
    return p;
  };
  _Float16* wqH   = (_Float16*)carve((size_t)1536 * 512 * 2);
  _Float16* woutH = (_Float16*)carve((size_t)512 * 512 * 2);
  _Float16* wfc1H = (_Float16*)carve((size_t)2048 * 512 * 2);
  _Float16* wfc2H = (_Float16*)carve((size_t)512 * 2048 * 2);
  // Shared act region, reused sequentially: xn -> attention o -> ln2h.
  _Float16* act512 = (_Float16*)carve((size_t)TOK * 512 * 2);
  // Tail region: qkv (TOKx1536), later reused for FC1 hidden (TOKx2048).
  _Float16* tail = (_Float16*)carve((size_t)TOK * 2048 * 2);
  _Float16* qkvh = tail;
  _Float16* h1h  = tail;

  // 1. f16 weights.
  cvt_kernel<<<(1536 * 512 + 255) / 256, 256, 0, stream>>>(qkv_w, wqH, 1536 * 512);
  cvt_kernel<<<(512 * 512 + 255) / 256, 256, 0, stream>>>(out_w, woutH, 512 * 512);
  cvt_kernel<<<(2048 * 512 + 255) / 256, 256, 0, stream>>>(fc1_w, wfc1H, 2048 * 512);
  cvt_kernel<<<(512 * 2048 + 255) / 256, 256, 0, stream>>>(fc2_w, wfc2H, 512 * 2048);

  // 2. LN1 + shift + window partition -> f16 windowed activations.
  ln_kernel<<<TOK, 256, 0, stream>>>(x, n1w, n1b, act512, 1);

  // 3. QKV projection (Q pre-scaled).
  {
    dim3 g(TOK / 128, 1536 / 128);
    gemm_f16<<<g, 256, 0, stream>>>(act512, wqH, qkv_b, TOK, 1536, 512,
                                    EPI_QKV, qkvh, nullptr, nullptr);
  }

  // 4. Windowed attention (o overwrites xn region).
  {
    dim3 g(Bn * NWw, NHh);
    attn_kernel<<<g, 128, 0, stream>>>(qkvh, rpb, act512);
  }

  // 5. Output projection + reverse shift scatter + shortcut -> d_out (= x_res).
  {
    dim3 g(TOK / 128, 512 / 128);
    gemm_f16<<<g, 256, 0, stream>>>(act512, woutH, out_b, TOK, 512, 512,
                                    EPI_OUTPROJ, nullptr, out, x);
  }

  // 6. LN2 (reads x_res in d_out; ln2h overwrites act region).
  ln_kernel<<<TOK, 256, 0, stream>>>(out, n2w, n2b, act512, 0);

  // 7. FC1 + exact GELU (hidden overwrites qkv region).
  {
    dim3 g(TOK / 128, 2048 / 128);
    gemm_f16<<<g, 256, 0, stream>>>(act512, wfc1H, fc1_b, TOK, 2048, 512,
                                    EPI_GELU, h1h, nullptr, nullptr);
  }

  // 8. FC2 + residual accumulate into d_out.
  {
    dim3 g(TOK / 128, 512 / 128);
    gemm_f16<<<g, 256, 0, stream>>>(h1h, wfc2H, fc2_b, TOK, 512, 2048,
                                    EPI_FC2, nullptr, out, nullptr);
  }

  (void)in_sizes; (void)n_in; (void)out_size; (void)ws_size;
}
